// SimilarityLayer_15118284882652
// MI455X (gfx1250) — matrix operand order repaired and checked by reading, not compiled, not run on hardware
//
#include <hip/hip_runtime.h>
#include <hip/hip_bf16.h>

// MI455X / gfx1250, wave32. f16 WMMA path (v_wmma_f32_16x16x32_f16).

typedef __attribute__((ext_vector_type(16))) _Float16 v16h;
typedef __attribute__((ext_vector_type(8)))  float    v8f;

#define E_   8
#define W_   5
#define C_   640
#define N_   49
#define Q_   75
#define M_   (W_ * N_)      // 245 rows (A side), padded to 256
#define QN_  (Q_ * N_)      // 3675 cols (B side), padded to 3712
#define MT_  16             // M tiles of 16  (256 rows)
#define NT_  232            // col tiles of 16, padded to multiple of 4 (3712 cols)
#define KT_  (C_ / 32)      // 20 K-steps of 32
#define TILE_ 512           // f16 elements per 16x32 operand tile (1 KB)

#define A_ELEMS ((size_t)E_ * MT_ * KT_ * TILE_)   // 1,310,720 f16  (2.6 MB)
#define B_ELEMS ((size_t)E_ * NT_ * KT_ * TILE_)   // 19,005,440 f16 (38 MB)

// ---------------------------------------------------------------------------
// Pack proto -> A operand register image.
// 16-bit A 16x32 layout: lane<16: row M=lane,   halves j: K = j (j<8), j+8  (j>=8)
//                        lane>=16: row M=lane-16, halves j: K = j+8 (j<8), j+16 (j>=8)
__global__ void pack_a(const float* __restrict__ proto, _Float16* __restrict__ Apk) {
    size_t id = (size_t)blockIdx.x * blockDim.x + threadIdx.x;
    if (id >= A_ELEMS) return;
    int    j    = (int)(id & 15);
    int    lane = (int)((id >> 4) & 31);
    size_t t    = id >> 9;                 // (e*MT + mt)*KT + kt
    int    kt   = (int)(t % KT_);
    size_t emt  = t / KT_;
    int    mt   = (int)(emt % MT_);
    int    e    = (int)(emt / MT_);
    int    hi   = lane >> 4;
    int    k_off = (j < 8) ? (j + (hi ? 8 : 0)) : (j + (hi ? 16 : 8));
    int    m = mt * 16 + (lane & 15);
    int    k = kt * 32 + k_off;
    float  v = 0.0f;
    if (m < M_) {
        int w = m / N_, n = m % N_;
        v = proto[(((size_t)e * W_ + w) * C_ + k) * N_ + n];
    }
    Apk[id] = (_Float16)v;
}

// Pack query -> B operand register image.
// 16-bit B 32x16 layout: lane<16: col N=lane,    halves j: K = j (j<8), j+8  (j>=8)
//                        lane>=16: col N=lane-16, halves j: K = j+8 (j<8), j+16 (j>=8)
__global__ void pack_b(const float* __restrict__ query, _Float16* __restrict__ Bpk) {
    size_t id = (size_t)blockIdx.x * blockDim.x + threadIdx.x;
    if (id >= B_ELEMS) return;
    int    j    = (int)(id & 15);
    int    lane = (int)((id >> 4) & 31);
    size_t t    = id >> 9;                 // (e*NT + nt)*KT + kt
    int    kt   = (int)(t % KT_);
    size_t ent  = t / KT_;
    int    nt   = (int)(ent % NT_);
    int    e    = (int)(ent / NT_);
    int    hi   = lane >> 4;
    int    k_off = (j < 8) ? (j + (hi ? 8 : 0)) : (j + (hi ? 16 : 8));
    int    qm = nt * 16 + (lane & 15);
    int    k  = kt * 32 + k_off;
    float  v  = 0.0f;
    if (qm < QN_) {
        int q = qm / N_, mm = qm % N_;
        v = query[(((size_t)e * Q_ + q) * C_ + k) * N_ + mm];
    }
    Bpk[id] = (_Float16)v;
}

// ---------------------------------------------------------------------------
// fp32 channel norms. One block per (e,w) / (e,q); lane <-> n so the
// N-contiguous innermost dim coalesces.
__global__ void norms_p(const float* __restrict__ proto, float* __restrict__ np) {
    int ew = blockIdx.x;                   // e*W + w
    int n  = threadIdx.x;
    if (n >= N_) return;
    const float* base = proto + (size_t)ew * C_ * N_ + n;
    float s = 0.0f;
    for (int c = 0; c < C_; ++c) { float x = base[(size_t)c * N_]; s += x * x; }
    np[ew * N_ + n] = sqrtf(s);
}

__global__ void norms_q(const float* __restrict__ query, float* __restrict__ nq) {
    int eq = blockIdx.x;                   // e*Q + q
    int n  = threadIdx.x;
    if (n >= N_) return;
    const float* base = query + (size_t)eq * C_ * N_ + n;
    float s = 0.0f;
    for (int c = 0; c < C_; ++c) { float x = base[(size_t)c * N_]; s += x * x; }
    nq[eq * N_ + n] = sqrtf(s);
}

// ---------------------------------------------------------------------------
// GEMM: one wave per 2(M-tiles) x 4(col-tiles) block of 16x16 output tiles.
// 8 waves per workgroup. K-loop: 20 steps x 32, 8 WMMAs per step.
__global__ void __launch_bounds__(256) sim_gemm(
        const _Float16* __restrict__ Apk, const _Float16* __restrict__ Bpk,
        const float* __restrict__ np, const float* __restrict__ nq,
        float* __restrict__ out) {
    const int NTQ = NT_ / 4;               // 58
    const int MTP = MT_ / 2;               // 8
    int wave = blockIdx.x * (blockDim.x >> 5) + (threadIdx.x >> 5);
    int lane = threadIdx.x & 31;
    int ntq  = wave % NTQ;
    int tmp  = wave / NTQ;
    int mtp  = tmp % MTP;
    int e    = tmp / MTP;                  // grid sized exactly; e < E_ always

    int mt0 = mtp * 2;
    int nt0 = ntq * 4;

    const _Float16* Abase = Apk + (((size_t)e * MT_ + mt0) * KT_) * TILE_ + (size_t)lane * 16;
    const _Float16* Bbase = Bpk + (((size_t)e * NT_ + nt0) * KT_) * TILE_ + (size_t)lane * 16;

    v8f acc[2][4];
    v8f zero = {};
#pragma unroll
    for (int i = 0; i < 2; ++i)
#pragma unroll
        for (int jn = 0; jn < 4; ++jn) acc[i][jn] = zero;

    for (int kt = 0; kt < KT_; ++kt) {
        // prefetch next K-step (emits global_prefetch; harmless speculative OOB)
        __builtin_prefetch(Abase + (size_t)(kt + 1) * TILE_, 0, 3);
        __builtin_prefetch(Bbase + (size_t)(kt + 1) * TILE_, 0, 3);

        v16h a0 = *(const v16h*)(Abase + (size_t)kt * TILE_);
        v16h a1 = *(const v16h*)(Abase + ((size_t)KT_ + kt) * TILE_);
#pragma unroll
        for (int jn = 0; jn < 4; ++jn) {
            v16h b = *(const v16h*)(Bbase + ((size_t)jn * KT_ + kt) * TILE_);
            acc[0][jn] = __builtin_amdgcn_wmma_f32_16x16x32_f16(
                false, a0, false, b, (short)0, acc[0][jn], false, false);
            acc[1][jn] = __builtin_amdgcn_wmma_f32_16x16x32_f16(
                false, a1, false, b, (short)0, acc[1][jn], false, false);
        }
    }

    // Epilogue: cosine normalization + scatter to [E,Q,W,N,N].
    // C/D layout: VGPR r, lane<16 -> (M=r, N=lane); lane>=16 -> (M=r+8, N=lane-16)
    int lanelo = lane & 15;
    int hi8    = (lane >> 4) * 8;
#pragma unroll
    for (int jn = 0; jn < 4; ++jn) {
        int qm = (nt0 + jn) * 16 + lanelo;
        if (qm >= QN_) continue;
        float nqv = nq[e * QN_ + qm];
        int q  = qm / N_;
        int mm = qm % N_;
        size_t obase = ((size_t)(e * Q_ + q)) * (W_ * N_ * N_);
#pragma unroll
        for (int i = 0; i < 2; ++i) {
            int mtb = (mt0 + i) * 16;
#pragma unroll
            for (int r = 0; r < 8; ++r) {
                int m = mtb + r + hi8;
                if (m < M_) {
                    float npv = np[e * M_ + m];
                    int w = m / N_, n = m % N_;
                    out[obase + (size_t)w * (N_ * N_) + n * N_ + mm] =
                        acc[i][jn][r] / (npv * nqv + 1e-8f);
                }
            }
        }
    }
}

// ---------------------------------------------------------------------------
extern "C" void kernel_launch(void* const* d_in, const int* in_sizes, int n_in,
                              void* d_out, int out_size, void* d_ws, size_t ws_size,
                              hipStream_t stream) {
    const float* proto = (const float*)d_in[0];   // [E,W,C,1,N] fp32
    const float* query = (const float*)d_in[1];   // [E,Q,C,1,N] fp32
    float*       out   = (float*)d_out;           // [E,Q,W,N,N] fp32

    // workspace carve-out (~40.8 MB): packed A | packed B | norms
    _Float16* Apk = (_Float16*)d_ws;
    _Float16* Bpk = Apk + A_ELEMS;
    float*    np  = (float*)(Bpk + B_ELEMS);
    float*    nq  = np + E_ * M_;

    pack_a<<<(unsigned)((A_ELEMS + 255) / 256), 256, 0, stream>>>(proto, Apk);
    pack_b<<<(unsigned)((B_ELEMS + 255) / 256), 256, 0, stream>>>(query, Bpk);
    norms_p<<<E_ * W_, 64, 0, stream>>>(proto, np);
    norms_q<<<E_ * Q_, 64, 0, stream>>>(query, nq);

    // 8 * (16/2) * (232/4) = 3712 waves -> 464 blocks of 8 waves
    int waves  = E_ * (MT_ / 2) * (NT_ / 4);
    sim_gemm<<<waves / 8, 256, 0, stream>>>(Apk, Bpk, np, nq, out);
}
